// Decoder_31516470018169
// MI455X (gfx1250) — matrix operand order repaired and checked by reading, not compiled, not run on hardware
//
#include <hip/hip_runtime.h>
#include <hip/hip_bf16.h>
#include <stdint.h>
#include <stddef.h>

#define UNITS    1024
#define HORIZON  24
#define FEATURES 128
#define BATCH    256
#define SEQ      512
#define NGATES   4096   // 4*UNITS

typedef __attribute__((ext_vector_type(16))) __bf16 v16bf;
typedef __attribute__((ext_vector_type(8)))  __bf16 v8bf;
typedef __attribute__((ext_vector_type(8)))  float  v8f;

__device__ __forceinline__ uint16_t f32_to_bf16_rne(float f) {
    uint32_t u = __float_as_uint(f);
    u += 0x7FFFu + ((u >> 16) & 1u);
    return (uint16_t)(u >> 16);
}

__device__ __forceinline__ float bf16_to_f32(uint16_t h) {
    return __uint_as_float(((uint32_t)h) << 16);
}

// v_exp_f32-based activations (single TRANS op instead of libm expansion)
#define LOG2E 1.44269504088896340736f
__device__ __forceinline__ float fast_sigmoid(float x) {
    return 1.0f / (1.0f + __builtin_amdgcn_exp2f(-LOG2E * x));
}
__device__ __forceinline__ float fast_tanh(float x) {
    return 2.0f / (1.0f + __builtin_amdgcn_exp2f(-2.0f * LOG2E * x)) - 1.0f;
}

// ---------------------------------------------------------------------------
// Prep: dst[n*K + k] = bf16(src1[k*4096 + n] (+ src2[k*4096 + n]))
// [K,4096] fp32 -> [4096,K] bf16 so each lane's B-fragment is one contiguous
// 32B K-run. src2 != null fuses cell_W + cell_U (decoder input == hidden).
// ---------------------------------------------------------------------------
__global__ void transpose_convert_kernel(const float* __restrict__ src1,
                                         const float* __restrict__ src2,
                                         uint16_t* __restrict__ dst, int K) {
    int idx = blockIdx.x * blockDim.x + threadIdx.x;
    if (idx >= K * NGATES) return;
    int n = idx / K;
    int k = idx - n * K;
    float v = src1[(size_t)k * NGATES + n];
    if (src2) v += src2[(size_t)k * NGATES + n];
    dst[idx] = f32_to_bf16_rne(v);
}

// Prep: flat fp32 -> bf16 (whole x tensor), 8 elems/thread, packed b128 store
__global__ void convert_bf16_kernel(const float* __restrict__ src,
                                    uint32_t* __restrict__ dst, int n8) {
    int i = blockIdx.x * blockDim.x + threadIdx.x;
    if (i >= n8) return;
    const float* s = src + (size_t)i * 8;
    uint32_t d[4];
    #pragma unroll
    for (int c = 0; c < 4; ++c) {
        uint32_t lo = __float_as_uint(s[2 * c]);
        uint32_t hh = __float_as_uint(s[2 * c + 1]);
        lo = (lo + (0x7FFFu + ((lo >> 16) & 1u))) >> 16;
        hh = (hh + (0x7FFFu + ((hh >> 16) & 1u))) & 0xFFFF0000u;
        d[c] = hh | lo;
    }
    uint32_t* p = dst + (size_t)i * 4;
    p[0] = d[0]; p[1] = d[1]; p[2] = d[2]; p[3] = d[3];
}

__global__ void zero_kernel_u32(uint32_t* __restrict__ p, int n) {
    int i = blockIdx.x * blockDim.x + threadIdx.x;
    if (i < n) p[i] = 0u;
}

// ---------------------------------------------------------------------------
// Barrier-free GEMM phase, A bf16 in global (L2-resident):
//   acc += A[m0:m0+32, :Kphase] * B[:Kphase, gate cols]
// A fragments: two contiguous 16B runs per lane per tile (adjacent, k = 16*(lane/16) + s, same order as B).
// B fragments: one contiguous 32B K-run per lane (BT is [4096][Kphase]).
// ---------------------------------------------------------------------------
__device__ __forceinline__ void mma_phase_bf16(
    const uint16_t* __restrict__ Abf, int lda,
    const uint16_t* __restrict__ BT, int Kphase,
    int m0, int nglob, int lane,
    v8f& acc0, v8f& acc1)
{
    const int l15  = lane & 15;
    const int hi   = lane >> 4;
    const int kb_a = hi * 16;
    const int kb_b = hi * 16;
    const uint16_t* arow0 = Abf + (size_t)(m0 + l15) * lda + kb_a;
    const uint16_t* arow1 = arow0 + (size_t)16 * lda;
    const uint16_t* brow  = BT + (size_t)(nglob + l15) * Kphase + kb_b;

    for (int k0 = 0; k0 < Kphase; k0 += 32) {
        v8bf a0lo = *(const v8bf*)(arow0 + k0);
        v8bf a0hi = *(const v8bf*)(arow0 + k0 + 8);
        v8bf a1lo = *(const v8bf*)(arow1 + k0);
        v8bf a1hi = *(const v8bf*)(arow1 + k0 + 8);
        v16bf b   = *(const v16bf*)(brow + k0);
        v16bf a0, a1;
        #pragma unroll
        for (int i = 0; i < 8; ++i) {
            a0[i] = a0lo[i]; a0[i + 8] = a0hi[i];
            a1[i] = a1lo[i]; a1[i + 8] = a1hi[i];
        }
        acc0 = __builtin_amdgcn_wmma_f32_16x16x32_bf16(false, a0, false, b,
                                                       (short)0, acc0, false, false);
        acc1 = __builtin_amdgcn_wmma_f32_16x16x32_bf16(false, a1, false, b,
                                                       (short)0, acc1, false, false);
        if (k0 + 32 < Kphase)
            __builtin_prefetch(brow + k0 + 32, 0, 3);   // global_prefetch_b8
    }
}

// ---------------------------------------------------------------------------
// One LSTM timestep, fully fused. Grid (256/32, 1024/16), 128 threads.
// Wave g computes gate g (i,f,g,o) for the same 16 hidden columns so gate
// math fuses through LDS; h is emitted directly as bf16 for the next step.
// Encoder: phase0 = x_t (K0=128) + phase1 = h (Ku=1024).
// Decoder: phase0 = h vs (cell_W + cell_U), Ku=0.
// ---------------------------------------------------------------------------
__global__ void __launch_bounds__(128)
lstm_step_kernel(const uint16_t* __restrict__ Abf0, int lda0,
                 const uint16_t* __restrict__ BT0, int K0,
                 const uint16_t* __restrict__ Hbf,
                 const uint16_t* __restrict__ UT, int Ku,
                 const float* __restrict__ bias,
                 float* __restrict__ C,
                 uint16_t* __restrict__ Hbf_out)
{
    __shared__ float zf[4][32][16];

    const int tid  = threadIdx.x;
    const int wave = tid >> 5;                 // gate 0..3
    const int lane = tid & 31;
    const int m0   = blockIdx.x * 32;
    const int n0   = blockIdx.y * 16;
    const int nglob = wave * UNITS + n0;

    v8f acc0 = {}; v8f acc1 = {};

    mma_phase_bf16(Abf0, lda0, BT0, K0, m0, nglob, lane, acc0, acc1);
    if (Ku > 0)
        mma_phase_bf16(Hbf, UNITS, UT, Ku, m0, nglob, lane, acc0, acc1);

    // accumulators + bias -> LDS (C/D layout: col = lane%16, lanes>=16 rows+8)
    const float bv   = bias[nglob + (lane & 15)];
    const int   colw = lane & 15;
    const int   rb   = (lane >> 4) ? 8 : 0;
    #pragma unroll
    for (int r = 0; r < 8; ++r) {
        zf[wave][rb + r][colw]      = acc0[r] + bv;
        zf[wave][16 + rb + r][colw] = acc1[r] + bv;
    }
    __syncthreads();

    // fused gates: c' = sig(f)*c + sig(i)*tanh(g); h' = sig(o)*tanh(c') -> bf16
    for (int e = tid; e < 32 * 16; e += 128) {
        int row = e >> 4, col = e & 15;
        float zi  = zf[0][row][col];
        float zff = zf[1][row][col];
        float zg  = zf[2][row][col];
        float zo  = zf[3][row][col];
        size_t idx = (size_t)(m0 + row) * UNITS + (n0 + col);
        float c_new = fast_sigmoid(zff) * C[idx] + fast_sigmoid(zi) * fast_tanh(zg);
        C[idx] = c_new;
        Hbf_out[idx] = f32_to_bf16_rne(fast_sigmoid(zo) * fast_tanh(c_new));
    }
}

// pred[b] = h[b,:] . dense_W + dense_b (bf16 h, fp32 weights)
__global__ void __launch_bounds__(128)
dense_kernel(const uint16_t* __restrict__ Hbf, const float* __restrict__ dW,
             const float* __restrict__ db, float* __restrict__ out, int step)
{
    __shared__ float red[128];
    const int b = blockIdx.x, tid = threadIdx.x;
    const uint16_t* hrow = Hbf + (size_t)b * UNITS;
    float s = 0.0f;
    for (int j = tid; j < UNITS; j += 128) s += bf16_to_f32(hrow[j]) * dW[j];
    red[tid] = s;
    __syncthreads();
    for (int off = 64; off > 0; off >>= 1) {
        if (tid < off) red[tid] += red[tid + off];
        __syncthreads();
    }
    if (tid == 0) out[(size_t)b * HORIZON + step] = red[0] + db[0];
}

// ---------------------------------------------------------------------------
extern "C" void kernel_launch(void* const* d_in, const int* in_sizes, int n_in,
                              void* d_out, int out_size, void* d_ws, size_t ws_size,
                              hipStream_t stream)
{
    const float* x       = (const float*)d_in[0];
    const float* enc_W   = (const float*)d_in[1];
    const float* enc_U   = (const float*)d_in[2];
    const float* enc_b   = (const float*)d_in[3];
    const float* cell_W  = (const float*)d_in[4];
    const float* cell_U  = (const float*)d_in[5];
    const float* cell_b  = (const float*)d_in[6];
    const float* dense_W = (const float*)d_in[7];
    const float* dense_b = (const float*)d_in[8];
    float* out = (float*)d_out;
    (void)in_sizes; (void)n_in; (void)out_size; (void)ws_size;

    // workspace carve (~51 MB: x_bf16 32MB + weights 17MB + state)
    char* ws = (char*)d_ws;
    auto carve = [&](size_t bytes) {
        char* p = ws; ws += (bytes + 255) & ~(size_t)255; return p;
    };
    uint16_t* Xbf    = (uint16_t*)carve((size_t)BATCH * SEQ * FEATURES * 2);
    uint16_t* WT_enc = (uint16_t*)carve((size_t)NGATES * FEATURES * 2);
    uint16_t* UT_enc = (uint16_t*)carve((size_t)NGATES * UNITS * 2);
    uint16_t* CombT  = (uint16_t*)carve((size_t)NGATES * UNITS * 2);  // cell_W + cell_U
    uint16_t* Hbf_a  = (uint16_t*)carve((size_t)BATCH * UNITS * 2);
    uint16_t* Hbf_b  = (uint16_t*)carve((size_t)BATCH * UNITS * 2);
    float*    C      = (float*)carve((size_t)BATCH * UNITS * 4);

    // --- prep: convert x, transpose/convert weights, zero h0/c0 ---
    {
        int n8 = BATCH * SEQ * FEATURES / 8;
        convert_bf16_kernel<<<(n8 + 255) / 256, 256, 0, stream>>>(x, (uint32_t*)Xbf, n8);
        int t1 = NGATES * FEATURES;
        transpose_convert_kernel<<<(t1 + 255) / 256, 256, 0, stream>>>(enc_W, nullptr, WT_enc, FEATURES);
        int t2 = NGATES * UNITS;
        transpose_convert_kernel<<<(t2 + 255) / 256, 256, 0, stream>>>(enc_U, nullptr, UT_enc, UNITS);
        transpose_convert_kernel<<<(t2 + 255) / 256, 256, 0, stream>>>(cell_W, cell_U, CombT, UNITS);
        int nzH = BATCH * UNITS / 2;
        zero_kernel_u32<<<(nzH + 255) / 256, 256, 0, stream>>>((uint32_t*)Hbf_a, nzH);
        int nzC = BATCH * UNITS;
        zero_kernel_u32<<<(nzC + 255) / 256, 256, 0, stream>>>((uint32_t*)C, nzC);
    }

    dim3 sgrid(BATCH / 32, UNITS / 16);   // 8 x 64
    uint16_t* Hcur  = Hbf_a;
    uint16_t* Hnext = Hbf_b;

    // --- encoder: 512 sequential steps (x row stride = SEQ*FEATURES) ---
    for (int t = 0; t < SEQ; ++t) {
        lstm_step_kernel<<<sgrid, 128, 0, stream>>>(
            Xbf + (size_t)t * FEATURES, SEQ * FEATURES, WT_enc, FEATURES,
            Hcur, UT_enc, UNITS, enc_b, C, Hnext);
        uint16_t* tmp = Hcur; Hcur = Hnext; Hnext = tmp;
    }

    // --- decoder: input == hidden, single GEMM vs combined (W+U) ---
    for (int s = 0; s < HORIZON; ++s) {
        lstm_step_kernel<<<sgrid, 128, 0, stream>>>(
            Hcur, UNITS, CombT, UNITS,
            Hcur, UT_enc, 0 /*skip U phase*/, cell_b, C, Hnext);
        dense_kernel<<<BATCH, 128, 0, stream>>>(Hnext, dense_W, dense_b, out, s);
        uint16_t* tmp = Hcur; Hcur = Hnext; Hnext = tmp;
    }
}
